// CrossFrameSimilarityRefiner_43928925504047
// MI455X (gfx1250) — compile-verified, matrix operand order checked by reading, not run on hardware
//
#include <hip/hip_runtime.h>

typedef __attribute__((ext_vector_type(16))) _Float16 v16h;
typedef __attribute__((ext_vector_type(8)))  float    v8f;

#define T_   16
#define B_   8
#define C_   256
#define P_   1024
#define KTOP 3
#define EPSF 1e-8f

// ---------------------------------------------------------------------------
// CDNA5 async global->LDS copy (16B per lane), tracked by ASYNCcnt.
// vDST operand = workgroup-relative LDS byte offset (generic LDS pointers keep
// the offset in the low 32 bits per the aperture mapping, ISA 10.2).
// ---------------------------------------------------------------------------
__device__ __forceinline__ void async_copy_b128(unsigned lds_off, const void* g) {
  asm volatile("global_load_async_to_lds_b128 %0, %1, off"
               :: "v"(lds_off), "v"(g)
               : "memory");
}

// ---------------------------------------------------------------------------
// Kernel 1: per (t,b,p) stats over C, plus f32->f16 TRANSPOSED feature copy
//   fhT[tb][p][c] = (f16) f[tb][c][p]   (so GEMM B-tiles are k-contiguous)
//   S[t,b,p] = (sum_c f) / max(||f||_2, eps)
//   M[t,b,p] = mean_c f / max(|f|, eps)
// Loads are lane-coalesced along p; f16 stores are packed 16B per 8 channels.
// ---------------------------------------------------------------------------
__global__ void cfsr_stats_kernel(const float* __restrict__ f,
                                  float* __restrict__ Sb,
                                  float* __restrict__ Mb,
                                  _Float16* __restrict__ fhT) {
  int g  = blockIdx.x * blockDim.x + threadIdx.x;   // 0 .. T*B*P-1
  int p  = g & (P_ - 1);
  int tb = g >> 10;
  const float* base = f + (size_t)tb * C_ * P_ + p;
  _Float16* hrow    = fhT + ((size_t)tb * P_ + p) * C_;   // contiguous in c
  float sumsq = 0.f, sum = 0.f, sm = 0.f;
  for (int c = 0; c < C_; c += 8) {
    union { _Float16 h[8]; uint4 u; } pk;
#pragma unroll
    for (int j = 0; j < 8; ++j) {
      float v = base[(size_t)(c + j) * P_];
      sumsq += v * v;
      sum   += v;
      sm    += v / fmaxf(fabsf(v), EPSF);
      pk.h[j] = (_Float16)v;
    }
    *(uint4*)(hrow + c) = pk.u;                      // 16B store, aligned
  }
  Sb[g] = sum / fmaxf(sqrtf(sumsq), EPSF);
  Mb[g] = sm * (1.0f / (float)C_);
}

// ---------------------------------------------------------------------------
// Kernel 2: W (f32) -> Wh (f16), 65536 elements.
// ---------------------------------------------------------------------------
__global__ void cfsr_wcvt_kernel(const float* __restrict__ W,
                                 _Float16* __restrict__ Wh) {
  int g = blockIdx.x * blockDim.x + threadIdx.x;
  Wh[g] = (_Float16)W[g];
}

// ---------------------------------------------------------------------------
// Kernel 3: scores[t,b,s] = (1/P) sum_p S[t,b,p] * M[s,b,p].  2048 dot-1024s.
// ---------------------------------------------------------------------------
__global__ void cfsr_scores_kernel(const float* __restrict__ Sb,
                                   const float* __restrict__ Mb,
                                   float* __restrict__ scores) {
  int g = blockIdx.x * blockDim.x + threadIdx.x;    // T*B*T = 2048
  int s = g & 15;
  int t = (g >> 4) & 15;
  int b = g >> 8;
  const float* srow = Sb + (size_t)(t * B_ + b) * P_;
  const float* mrow = Mb + (size_t)(s * B_ + b) * P_;
  float dot = 0.f;
#pragma unroll 4
  for (int p = 0; p < P_; ++p) dot += srow[p] * mrow[p];
  scores[((t * B_ + b) << 4) + s] = dot * (1.0f / (float)P_);
}

// ---------------------------------------------------------------------------
// Kernel 4: top-3 per (t,b) over the 15 off-diagonal (compressed) scores.
// Keeps the reference's "bug": the compressed index (0..14) IS the frame id.
// ---------------------------------------------------------------------------
__global__ void cfsr_topk_kernel(const float* __restrict__ scores,
                                 int* __restrict__ idx) {
  int g = threadIdx.x;                              // 128 threads, 1 block
  if (g >= T_ * B_) return;
  int t = g / B_;
  const float* sc = scores + (g << 4);
  float v[T_ - 1];
#pragma unroll
  for (int k = 0; k < T_ - 1; ++k) v[k] = sc[k < t ? k : k + 1];
#pragma unroll
  for (int j = 0; j < KTOP; ++j) {
    int bi = 0; float bv = v[0];
#pragma unroll
    for (int k = 1; k < T_ - 1; ++k)
      if (v[k] > bv) { bv = v[k]; bi = k; }
    idx[g * KTOP + j] = bi;
    v[bi] = -3.4e38f;
  }
}

// ---------------------------------------------------------------------------
// Kernel 5: per-frame projection GEMM  Y[tb,d,p] = sum_c Wh[d,c] * fhT[tb,p,c]
//
// Block tile 256(d) x 64(p), K stepped by 32.  8 waves; wave w owns two
// d-subtiles (rows w*32 .. w*32+31) and all 4 p-subtiles -> 8 accumulators,
// 8 x v_wmma_f32_16x16x32_f16 per K step, each B-fragment reused twice.
//
// Staging: double-buffered GLOBAL_LOAD_ASYNC_TO_LDS_B128 (ASYNCcnt pipeline).
// Both tiles are [row][k] with k contiguous; LDS row stride 40 halfs (80 B,
// 16B multiple for b128 LDS writes; 20-bank stride -> near conflict-free
// fragment gathers). Per wave per step: 4 A + 1 B async ops -> wait <=5 while
// the previous tile is being consumed.
//
// Fragment gathers follow the ISA 16-bit 16x32 layout: lane l -> row l&15,
// K-group (l>>4)*8; VGPR j holds the contiguous K pair {2j,2j+1} (+16, j>=4).
// Epilogue uses the f32 C/D layout (VGPR r -> M=r+8*lane[4], N=lane&15) and
// stores coalesced along p.
// ---------------------------------------------------------------------------
#define PTILE 64
#define KC    32
#define PADK  40            // halfs per LDS row: 80 B

__global__ void cfsr_gemm_kernel(const _Float16* __restrict__ Wh,
                                 const _Float16* __restrict__ fhT,
                                 float* __restrict__ Y) {
  __shared__ alignas(16) _Float16 As[2][C_ * PADK];     // 2 x 20 KiB
  __shared__ alignas(16) _Float16 Bs[2][PTILE * PADK];  // 2 x  5 KiB

  int bid = blockIdx.x;
  int pt  = bid & 15;                   // P_/PTILE = 16
  int tb  = bid >> 4;                   // 0..127
  int p0  = pt * PTILE;
  const _Float16* fbase = fhT + ((size_t)tb * P_ + p0) * C_;  // row p: +C_

  int tid = threadIdx.x;
  int l   = tid & 31;                   // lane in wave32
  int w   = tid >> 5;                   // wave id 0..7
  int m   = l & 15;
  int kb  = (l >> 4) << 3;

  auto prefetch = [&](int buf, int c0) {
    unsigned ab = (unsigned)(uintptr_t)&As[buf][0];
    unsigned bb = (unsigned)(uintptr_t)&Bs[buf][0];
    // A: 256 rows x 32 halfs = 1024 b128 chunks, 4 per thread
#pragma unroll
    for (int i = 0; i < 4; ++i) {
      int e = i * 256 + tid;
      int row = e >> 2, c16 = e & 3;
      async_copy_b128(ab + (unsigned)(row * PADK + c16 * 8) * 2u,
                      Wh + (size_t)row * C_ + c0 + c16 * 8);
    }
    // B: 64 rows x 32 halfs = 256 chunks, 1 per thread
    {
      int row = tid >> 2, c16 = tid & 3;
      async_copy_b128(bb + (unsigned)(row * PADK + c16 * 8) * 2u,
                      fbase + (size_t)row * C_ + c0 + c16 * 8);
    }
  };

  v8f acc[2][4] = {};

  prefetch(0, 0);

  for (int step = 0; step < C_ / KC; ++step) {
    int cur = step & 1;
    if (step < C_ / KC - 1) {
      prefetch(cur ^ 1, (step + 1) * KC);               // ASYNCcnt += 5
      asm volatile("s_wait_asynccnt 0x5" ::: "memory"); // current tile landed
    } else {
      asm volatile("s_wait_asynccnt 0x0" ::: "memory");
    }
    __syncthreads();                                    // all waves' tile ready

    // A fragments for this wave's two d-subtiles
    v16h af[2];
#pragma unroll
    for (int s = 0; s < 2; ++s) {
      const _Float16* ap = &As[cur][(w * 32 + s * 16 + m) * PADK];
#pragma unroll
      for (int j = 0; j < 8; ++j) {
        int K2 = ((j < 4) ? (2 * j) : (16 + 2 * (j - 4))) + kb;
        af[s][2 * j]     = ap[K2];
        af[s][2 * j + 1] = ap[K2 + 1];
      }
    }
    // 4 p-subtiles, each B fragment feeds 2 WMMAs
#pragma unroll
    for (int ps = 0; ps < 4; ++ps) {
      v16h bf;
      const _Float16* bp = &Bs[cur][(ps * 16 + m) * PADK];
#pragma unroll
      for (int j = 0; j < 8; ++j) {
        int K2 = ((j < 4) ? (2 * j) : (16 + 2 * (j - 4))) + kb;
        bf[2 * j]     = bp[K2];
        bf[2 * j + 1] = bp[K2 + 1];
      }
      acc[0][ps] = __builtin_amdgcn_wmma_f32_16x16x32_f16(
          false, af[0], false, bf, (short)0, acc[0][ps], false, false);
      acc[1][ps] = __builtin_amdgcn_wmma_f32_16x16x32_f16(
          false, af[1], false, bf, (short)0, acc[1][ps], false, false);
    }
    __syncthreads();                    // everyone done reading before rewrite
  }

  // Epilogue: coalesced f32 stores along p
  float* ybase = Y + (size_t)tb * C_ * P_;
  int rowoff = (l >> 4) << 3;
#pragma unroll
  for (int s = 0; s < 2; ++s) {
#pragma unroll
    for (int ps = 0; ps < 4; ++ps) {
      int p = p0 + ps * 16 + m;
#pragma unroll
      for (int r = 0; r < 8; ++r) {
        int d = w * 32 + s * 16 + rowoff + r;
        ybase[(size_t)d * P_ + p] = acc[s][ps][r];
      }
    }
  }
}

// ---------------------------------------------------------------------------
// Kernel 6: out[t,b,d,p] = mean_k Y[idx_k, b, d, p] + bias[d], float4 vectors.
// ---------------------------------------------------------------------------
__global__ void cfsr_gather_kernel(const float4* __restrict__ Y4,
                                   const int* __restrict__ idx,
                                   const float* __restrict__ bias,
                                   float4* __restrict__ out4) {
  size_t g   = (size_t)blockIdx.x * blockDim.x + threadIdx.x; // T*B*C*P/4
  int p4     = (int)(g & 255);           // P_/4 = 256
  size_t r8  = g >> 8;
  int d      = (int)(r8 & 255);
  int tb     = (int)(r8 >> 8);
  int b      = tb & (B_ - 1);
  const int* id = idx + tb * KTOP;
  float4 r; r.x = 0.f; r.y = 0.f; r.z = 0.f; r.w = 0.f;
#pragma unroll
  for (int k = 0; k < KTOP; ++k) {
    size_t src = (((size_t)(id[k] * B_ + b) * C_ + d) << 8) + p4;
    float4 y = Y4[src];
    r.x += y.x; r.y += y.y; r.z += y.z; r.w += y.w;
  }
  float bb = bias[d];
  const float inv = 1.0f / (float)KTOP;
  r.x = r.x * inv + bb;
  r.y = r.y * inv + bb;
  r.z = r.z * inv + bb;
  r.w = r.w * inv + bb;
  out4[g] = r;
}

// ---------------------------------------------------------------------------
// Launch
// ---------------------------------------------------------------------------
extern "C" void kernel_launch(void* const* d_in, const int* in_sizes, int n_in,
                              void* d_out, int out_size, void* d_ws, size_t ws_size,
                              hipStream_t stream) {
  const float* f    = (const float*)d_in[0];   // [T,B,C,H,W]
  const float* W    = (const float*)d_in[1];   // [C,C]
  const float* bias = (const float*)d_in[2];   // [C]
  (void)d_in[3]; (void)in_sizes; (void)n_in; (void)out_size; (void)ws_size;

  // Workspace layout (bytes)
  char* ws = (char*)d_ws;
  const size_t Y_BYTES  = (size_t)T_ * B_ * C_ * P_ * sizeof(float);     // 128 MiB
  const size_t FH_BYTES = (size_t)T_ * B_ * C_ * P_ * sizeof(_Float16);  //  64 MiB
  const size_t SM_BYTES = (size_t)T_ * B_ * P_ * sizeof(float);          // 512 KiB
  float*    Yb     = (float*)ws;
  _Float16* fhT    = (_Float16*)(ws + Y_BYTES);
  float*    Sb     = (float*)(ws + Y_BYTES + FH_BYTES);
  float*    Mb     = (float*)(ws + Y_BYTES + FH_BYTES + SM_BYTES);
  float*    scores = (float*)(ws + Y_BYTES + FH_BYTES + 2 * SM_BYTES);
  int*      idx    = (int*)(ws + Y_BYTES + FH_BYTES + 2 * SM_BYTES + 8192);
  _Float16* Wh     = (_Float16*)(ws + Y_BYTES + FH_BYTES + 2 * SM_BYTES + 8192 + 4096);

  // 1. stats + transposed f16 feature copy
  cfsr_stats_kernel<<<(T_ * B_ * P_) / 256, 256, 0, stream>>>(f, Sb, Mb, fhT);
  // 2. W -> f16
  cfsr_wcvt_kernel<<<(C_ * C_) / 256, 256, 0, stream>>>(W, Wh);
  // 3. similarity scores
  cfsr_scores_kernel<<<(T_ * B_ * T_) / 256, 256, 0, stream>>>(Sb, Mb, scores);
  // 4. top-k
  cfsr_topk_kernel<<<1, 128, 0, stream>>>(scores, idx);
  // 5. per-frame projection GEMM (WMMA + async double-buffered LDS staging)
  cfsr_gemm_kernel<<<(T_ * B_) * (P_ / PTILE), 256, 0, stream>>>(Wh, fhT, Yb);
  // 6. gather / average / bias
  cfsr_gather_kernel<<<(T_ * B_ * C_ * P_ / 4) / 256, 256, 0, stream>>>(
      (const float4*)Yb, idx, bias, (float4*)d_out);
}
